// PreNormCrossScaleBlock_36266703847805
// MI455X (gfx1250) — matrix operand order repaired and checked by reading, not compiled, not run on hardware
//
#include <hip/hip_runtime.h>
#include <hip/hip_bf16.h>

typedef _Float16 f16;
typedef __attribute__((ext_vector_type(16))) _Float16 v16h;
typedef __attribute__((ext_vector_type(8)))  _Float16 v8h;
typedef __attribute__((ext_vector_type(8)))  float    v8f;

// Problem constants
constexpr int NB   = 4;     // batch
constexpr int LQ   = 1024;
constexpr int LKV  = 2048;
constexpr int DM   = 1024;  // model dim
constexpr int NH   = 16;
constexpr int HD   = 64;
constexpr int DFF  = 4096;

// ---------------------------------------------------------------------------
// Shared epilogue math
// ---------------------------------------------------------------------------
__device__ __forceinline__ float ep_apply(float v, int ep, float g, float res) {
  if (ep == 1) v = 0.5f * v * (1.f + erff(v * 0.70710678118654752f));
  if (ep == 2) v = res + v * g;
  return v;
}

// ---------------------------------------------------------------------------
// BIG WMMA GEMM: block 128 thr = 4 waves (2x2), block tile 128x128,
// wave tile 64x64 (4x4 WMMA tiles), K step 32. Requires M%128==0, N%128==0.
// C[M,N] = alpha * A[M,K] @ Bt[K,N] + bias[N] (+ epilogue)
// ep: 0=none, 1=exact GELU, 2=residual + clip(gamma,±10) scale.
// ---------------------------------------------------------------------------
__global__ __launch_bounds__(128)
void gemm_wmma_big_k(const f16* __restrict__ A, const f16* __restrict__ Bt,
                     const float* __restrict__ bias, void* __restrict__ Cv,
                     const float* __restrict__ resid, const float* __restrict__ gamma,
                     int K, int lda, int ldb, int ldc,
                     long long Abs, long long Bbs, long long Cbs,
                     float alpha, int ep, int outF16)
{
  const int lane = threadIdx.x & 31;
  const int wave = threadIdx.x >> 5;
  const int m0 = blockIdx.y * 128 + (wave >> 1) * 64;
  const int n0 = blockIdx.x * 128 + (wave & 1) * 64;
  const int r  = lane & 15;
  const int hg = lane >> 4;
  const f16* Ap = A  + (size_t)blockIdx.z * (size_t)Abs;
  const f16* Bp = Bt + (size_t)blockIdx.z * (size_t)Bbs;

  v8f acc[4][4] = {};

  for (int k0 = 0; k0 < K; k0 += 32) {
    v16h a[4], b[4];
#pragma unroll
    for (int i = 0; i < 4; ++i) {
      // ISA 16-bit A layout (16x32): lane r/hg holds K chunks {hg*8..} and {16+hg*8..}
      const f16* p = Ap + (size_t)(m0 + i * 16 + r) * (size_t)lda + k0 + hg * 8;
      v8h c0 = *(const v8h*)p;
      v8h c1 = *(const v8h*)(p + 16);
#pragma unroll
      for (int e = 0; e < 8; ++e) { a[i][e] = c0[e]; a[i][e + 8] = c1[e]; }
    }
#pragma unroll
    for (int j = 0; j < 4; ++j)  // B operand: lane r/hg = N column, K chunks {hg*8..} and {16+hg*8..}
#pragma unroll
      for (int s = 0; s < 16; ++s)
        b[j][s] = Bp[(size_t)(k0 + hg * 8 + (s & 7) + (s >> 3) * 16) * (size_t)ldb + n0 + j * 16 + r];
#pragma unroll
    for (int i = 0; i < 4; ++i)
#pragma unroll
      for (int j = 0; j < 4; ++j)
        acc[i][j] = __builtin_amdgcn_wmma_f32_16x16x32_f16(
            false, a[i], false, b[j], (short)0, acc[i][j], false, false);
  }

  // Epilogue. C layout: row = m0+i*16+hg*8+e, col = n0+j*16+r
#pragma unroll
  for (int j = 0; j < 4; ++j) {
    const int col = n0 + j * 16 + r;
    const float bv = bias ? bias[col] : 0.f;
    float g = 0.f;
    if (ep == 2) g = fminf(fmaxf(gamma[col], -10.f), 10.f);
#pragma unroll
    for (int i = 0; i < 4; ++i) {
#pragma unroll
      for (int e = 0; e < 8; ++e) {
        const int row = m0 + i * 16 + hg * 8 + e;
        float res = 0.f;
        if (ep == 2) res = resid[(size_t)row * (size_t)ldc + col];
        float v = ep_apply(acc[i][j][e] * alpha + bv, ep, g, res);
        const size_t off = (size_t)blockIdx.z * (size_t)Cbs + (size_t)row * (size_t)ldc + col;
        if (outF16) ((f16*)Cv)[off] = (f16)v;
        else        ((float*)Cv)[off] = v;
      }
    }
  }
}

// ---------------------------------------------------------------------------
// SMALL WMMA GEMM: block 128 thr = 4 waves (2x2), block tile 64x64,
// wave tile 32x32. Requires M%64==0, N%64==0. Used for ctx (N=HD=64).
// ---------------------------------------------------------------------------
__global__ __launch_bounds__(128)
void gemm_wmma_small_k(const f16* __restrict__ A, const f16* __restrict__ Bt,
                       const float* __restrict__ bias, void* __restrict__ Cv,
                       int K, int lda, int ldb, int ldc,
                       long long Abs, long long Bbs, long long Cbs,
                       float alpha, int outF16)
{
  const int lane = threadIdx.x & 31;
  const int wave = threadIdx.x >> 5;
  const int m0 = blockIdx.y * 64 + (wave >> 1) * 32;
  const int n0 = blockIdx.x * 64 + (wave & 1) * 32;
  const int r  = lane & 15;
  const int hg = lane >> 4;
  const f16* Ap = A  + (size_t)blockIdx.z * (size_t)Abs;
  const f16* Bp = Bt + (size_t)blockIdx.z * (size_t)Bbs;

  v8f acc[2][2] = {};

  for (int k0 = 0; k0 < K; k0 += 32) {
    v16h a[2], b[2];
#pragma unroll
    for (int i = 0; i < 2; ++i) {
      const f16* p = Ap + (size_t)(m0 + i * 16 + r) * (size_t)lda + k0 + hg * 8;
      v8h c0 = *(const v8h*)p;
      v8h c1 = *(const v8h*)(p + 16);
#pragma unroll
      for (int e = 0; e < 8; ++e) { a[i][e] = c0[e]; a[i][e + 8] = c1[e]; }
    }
#pragma unroll
    for (int j = 0; j < 2; ++j)
#pragma unroll
      for (int s = 0; s < 16; ++s)
        b[j][s] = Bp[(size_t)(k0 + hg * 8 + (s & 7) + (s >> 3) * 16) * (size_t)ldb + n0 + j * 16 + r];
#pragma unroll
    for (int i = 0; i < 2; ++i)
#pragma unroll
      for (int j = 0; j < 2; ++j)
        acc[i][j] = __builtin_amdgcn_wmma_f32_16x16x32_f16(
            false, a[i], false, b[j], (short)0, acc[i][j], false, false);
  }

#pragma unroll
  for (int j = 0; j < 2; ++j) {
    const int col = n0 + j * 16 + r;
    const float bv = bias ? bias[col] : 0.f;
#pragma unroll
    for (int i = 0; i < 2; ++i) {
#pragma unroll
      for (int e = 0; e < 8; ++e) {
        const int row = m0 + i * 16 + hg * 8 + e;
        const float v = acc[i][j][e] * alpha + bv;
        const size_t off = (size_t)blockIdx.z * (size_t)Cbs + (size_t)row * (size_t)ldc + col;
        if (outF16) ((f16*)Cv)[off] = (f16)v;
        else        ((float*)Cv)[off] = v;
      }
    }
  }
}

// ---------------------------------------------------------------------------
// Block reduction helpers (256 threads)
// ---------------------------------------------------------------------------
__device__ __forceinline__ float block_sum256(float v, float* buf) {
  buf[threadIdx.x] = v; __syncthreads();
  for (int o = 128; o > 0; o >>= 1) {
    if ((int)threadIdx.x < o) buf[threadIdx.x] += buf[threadIdx.x + o];
    __syncthreads();
  }
  float r = buf[0]; __syncthreads();
  return r;
}
__device__ __forceinline__ float block_max256(float v, float* buf) {
  buf[threadIdx.x] = v; __syncthreads();
  for (int o = 128; o > 0; o >>= 1) {
    if ((int)threadIdx.x < o) buf[threadIdx.x] = fmaxf(buf[threadIdx.x], buf[threadIdx.x + o]);
    __syncthreads();
  }
  float r = buf[0]; __syncthreads();
  return r;
}

// ---------------------------------------------------------------------------
// LayerNorm over D=1024, f32 in -> f16 out, with row remapping (handles concat)
// dst_row = (row/src_rpb)*dst_rpb + dst_base + row%src_rpb
// ---------------------------------------------------------------------------
__global__ __launch_bounds__(256)
void ln_rows_k(const float* __restrict__ src, f16* __restrict__ dst,
               const float* __restrict__ g, const float* __restrict__ bta,
               int src_rpb, int dst_rpb, int dst_base)
{
  const int row = blockIdx.x;
  const int bi = row / src_rpb, ri = row % src_rpb;
  const float* x = src + (size_t)row * DM;
  f16* y = dst + ((size_t)bi * dst_rpb + dst_base + ri) * DM;
  __shared__ float buf[256];
  float v[4];
  float s = 0.f, s2 = 0.f;
#pragma unroll
  for (int t = 0; t < 4; ++t) {
    v[t] = x[threadIdx.x + t * 256];
    s += v[t]; s2 += v[t] * v[t];
  }
  s  = block_sum256(s, buf);
  s2 = block_sum256(s2, buf);
  const float mu = s * (1.f / DM);
  const float var = s2 * (1.f / DM) - mu * mu;
  const float rs = rsqrtf(var + 1e-5f);
#pragma unroll
  for (int t = 0; t < 4; ++t) {
    const int c = threadIdx.x + t * 256;
    y[c] = (f16)((v[t] - mu) * rs * g[c] + bta[c]);
  }
}

// ---------------------------------------------------------------------------
// Weight transpose + cast: W[N,K] f32  ->  Wt[K,N] f16
// ---------------------------------------------------------------------------
__global__ __launch_bounds__(256)
void transpose_w_k(const float* __restrict__ W, f16* __restrict__ Wt, int N, int K)
{
  __shared__ float t[32][33];
  const int k0 = blockIdx.x * 32, n0 = blockIdx.y * 32;
  const int tx = threadIdx.x, ty = threadIdx.y;   // (32, 8)
#pragma unroll
  for (int s = 0; s < 32; s += 8)
    t[ty + s][tx] = W[(size_t)(n0 + ty + s) * K + k0 + tx];
  __syncthreads();
#pragma unroll
  for (int s = 0; s < 32; s += 8)
    Wt[(size_t)(k0 + ty + s) * N + n0 + tx] = (f16)t[tx][ty + s];
}

// K-activation transpose: Kb[b*LKV + l, c] f16 -> Kt[b, c, l] f16
__global__ __launch_bounds__(256)
void transpose_k_k(const f16* __restrict__ Kb, f16* __restrict__ Kt)
{
  const int b = blockIdx.z;
  const f16* src = Kb + (size_t)b * LKV * DM;   // [LKV, DM]
  f16* dst = Kt + (size_t)b * DM * LKV;         // [DM, LKV]
  __shared__ f16 t[32][33];
  const int c0 = blockIdx.x * 32;  // column of src
  const int l0 = blockIdx.y * 32;  // row of src
  const int tx = threadIdx.x, ty = threadIdx.y;
#pragma unroll
  for (int s = 0; s < 32; s += 8)
    t[ty + s][tx] = src[(size_t)(l0 + ty + s) * DM + c0 + tx];
  __syncthreads();
#pragma unroll
  for (int s = 0; s < 32; s += 8)
    dst[(size_t)(c0 + ty + s) * LKV + l0 + tx] = t[tx][ty + s];
}

// ---------------------------------------------------------------------------
// Row softmax over LKV=2048 columns: in-place f32 (for head-mean output) and
// an f16 copy (A-operand of the ctx GEMM, keeps v_cvt out of the hot loop).
// ---------------------------------------------------------------------------
__global__ __launch_bounds__(256)
void softmax_rows_k(float* __restrict__ S, f16* __restrict__ P)
{
  const size_t base = (size_t)blockIdx.x * LKV;
  float* x = S + base;
  f16*  p = P + base;
  __shared__ float buf[256];
  float v[8];
  float m = -3.0e38f;
#pragma unroll
  for (int t = 0; t < 8; ++t) { v[t] = x[threadIdx.x + t * 256]; m = fmaxf(m, v[t]); }
  m = block_max256(m, buf);
  float s = 0.f;
#pragma unroll
  for (int t = 0; t < 8; ++t) { v[t] = expf(v[t] - m); s += v[t]; }
  s = block_sum256(s, buf);
  const float inv = 1.f / s;
#pragma unroll
  for (int t = 0; t < 8; ++t) {
    const int c = threadIdx.x + t * 256;
    const float pv = v[t] * inv;
    x[c] = pv;
    p[c] = (f16)pv;
  }
}

// Mean over the 16 heads of one batch slice: out[q,kv] = (1/16) sum_h S[h,q,kv]
__global__ __launch_bounds__(256)
void mean_heads_k(const float* __restrict__ S, float* __restrict__ out)
{
  const size_t idx = (size_t)blockIdx.x * 256 + threadIdx.x;
  float s = 0.f;
#pragma unroll
  for (int h = 0; h < NH; ++h) s += S[(size_t)h * ((size_t)LQ * LKV) + idx];
  out[idx] = s * (1.f / NH);
}

// ---------------------------------------------------------------------------
// Host launchers
// ---------------------------------------------------------------------------
static inline void gemm_big(hipStream_t st, const f16* A, const f16* Bt,
                            const float* bias, void* C, bool cf16,
                            const float* resid, const float* gamma,
                            int M, int N, int K, int lda, int ldb, int ldc,
                            long long Abs, long long Bbs, long long Cbs,
                            float alpha, int ep, int batch)
{
  dim3 g(N / 128, M / 128, batch), b(128);
  gemm_wmma_big_k<<<g, b, 0, st>>>(A, Bt, bias, C, resid, gamma, K,
                                   lda, ldb, ldc, Abs, Bbs, Cbs, alpha, ep, cf16 ? 1 : 0);
}

extern "C" void kernel_launch(void* const* d_in, const int* in_sizes, int n_in,
                              void* d_out, int out_size, void* d_ws, size_t ws_size,
                              hipStream_t stream)
{
  (void)in_sizes; (void)n_in; (void)out_size; (void)ws_size;
  const float* tight  = (const float*)d_in[0];
  const float* ctx_t  = (const float*)d_in[1];
  const float* nbh_t  = (const float*)d_in[2];
  const float* ln_q_g = (const float*)d_in[3];
  const float* ln_q_b = (const float*)d_in[4];
  const float* ln_kv_g= (const float*)d_in[5];
  const float* ln_kv_b= (const float*)d_in[6];
  const float* in_w   = (const float*)d_in[7];   // [3D, D]
  const float* in_b   = (const float*)d_in[8];   // [3D]
  const float* out_w  = (const float*)d_in[9];   // [D, D]
  const float* out_b  = (const float*)d_in[10];  // [D]
  const float* cgam   = (const float*)d_in[11];  // [D]
  const float* ln_f_g = (const float*)d_in[12];
  const float* ln_f_b = (const float*)d_in[13];
  const float* w1     = (const float*)d_in[14];  // [4D, D]
  const float* b1     = (const float*)d_in[15];
  const float* w2     = (const float*)d_in[16];  // [D, 4D]
  const float* b2     = (const float*)d_in[17];
  const float* fgam   = (const float*)d_in[18];
  float* out_x    = (float*)d_out;                                   // [NB*LQ, DM]
  float* out_mean = (float*)d_out + (size_t)NB * LQ * DM;            // [NB, LQ, LKV]

  // Workspace carve-up (256B aligned bump allocator)
  char* wp = (char*)d_ws;
  auto alloc = [&](size_t bytes) -> void* {
    void* p = wp; wp += (bytes + 255) & ~(size_t)255; return p;
  };
  const size_t MQ = (size_t)NB * LQ;      // 4096 rows
  const size_t MK = (size_t)NB * LKV;     // 8192 rows
  f16*  q_in  = (f16*) alloc(MQ * DM * 2);
  f16*  kv_in = (f16*) alloc(MK * DM * 2);
  f16*  WtIn  = (f16*) alloc((size_t)DM * 3 * DM * 2);   // [1024, 3072]
  f16*  WtOut = (f16*) alloc((size_t)DM * DM * 2);       // [1024, 1024]
  f16*  WtW1  = (f16*) alloc((size_t)DM * DFF * 2);      // [1024, 4096]
  f16*  WtW2  = (f16*) alloc((size_t)DFF * DM * 2);      // [4096, 1024]
  f16*  Qb    = (f16*) alloc(MQ * DM * 2);
  f16*  Kb    = (f16*) alloc(MK * DM * 2);
  f16*  Vb    = (f16*) alloc(MK * DM * 2);
  f16*  Kt    = (f16*) alloc((size_t)NB * DM * LKV * 2); // [b][h*HD][LKV]
  f16*  ctxb  = (f16*) alloc(MQ * DM * 2);
  f16*  lnf   = (f16*) alloc(MQ * DM * 2);
  f16*  h1    = (f16*) alloc(MQ * DFF * 2);
  float* Sbuf = (float*)alloc((size_t)NH * LQ * LKV * 4); // one batch of heads (f32)
  f16*  Pbuf  = (f16*) alloc((size_t)NH * LQ * LKV * 2);  // one batch of heads (f16)
  float* xbuf = (float*)alloc(MQ * DM * 4);

  // 1) Weight transposes (f32 -> f16, [N,K] -> [K,N])
  {
    dim3 blk(32, 8);
    transpose_w_k<<<dim3(DM/32, 3*DM/32), blk, 0, stream>>>(in_w,  WtIn,  3*DM, DM);
    transpose_w_k<<<dim3(DM/32,   DM/32), blk, 0, stream>>>(out_w, WtOut,   DM, DM);
    transpose_w_k<<<dim3(DM/32,  DFF/32), blk, 0, stream>>>(w1,    WtW1,   DFF, DM);
    transpose_w_k<<<dim3(DFF/32,  DM/32), blk, 0, stream>>>(w2,    WtW2,    DM, DFF);
  }

  // 2) LayerNorms -> f16 activations (kv concat via row remap)
  ln_rows_k<<<MQ, 256, 0, stream>>>(tight, q_in,  ln_q_g,  ln_q_b,  LQ, LQ,  0);
  ln_rows_k<<<MQ, 256, 0, stream>>>(ctx_t, kv_in, ln_kv_g, ln_kv_b, LQ, LKV, 0);
  ln_rows_k<<<MQ, 256, 0, stream>>>(nbh_t, kv_in, ln_kv_g, ln_kv_b, LQ, LKV, LQ);

  // 3) QKV projections (f16 out)
  gemm_big(stream, q_in,  WtIn,          in_b,          Qb, true, nullptr, nullptr,
           (int)MQ, DM, DM, DM, 3*DM, DM, 0, 0, 0, 1.f, 0, 1);
  gemm_big(stream, kv_in, WtIn +   DM,   in_b +   DM,   Kb, true, nullptr, nullptr,
           (int)MK, DM, DM, DM, 3*DM, DM, 0, 0, 0, 1.f, 0, 1);
  gemm_big(stream, kv_in, WtIn + 2*DM,   in_b + 2*DM,   Vb, true, nullptr, nullptr,
           (int)MK, DM, DM, DM, 3*DM, DM, 0, 0, 0, 1.f, 0, 1);

  // 4) K transpose per batch: Kt[b, h*HD+hd, l]
  transpose_k_k<<<dim3(DM/32, LKV/32, NB), dim3(32, 8), 0, stream>>>(Kb, Kt);

  // 5) Attention, batched over heads, serialized over batch (Sbuf/Pbuf reused)
  for (int b = 0; b < NB; ++b) {
    // scores: S[h,q,l] = (Q_h @ Kt_h) / 8    (f32 out)
    gemm_big(stream, Qb + (size_t)b * LQ * DM,
             Kt + (size_t)b * DM * LKV, nullptr, Sbuf, false, nullptr, nullptr,
             LQ, LKV, HD, DM, LKV, LKV,
             /*Abs*/ HD, /*Bbs*/ (long long)HD * LKV, /*Cbs*/ (long long)LQ * LKV,
             0.125f, 0, NH);
    // softmax rows: f32 in-place + f16 copy
    softmax_rows_k<<<NH * LQ, 256, 0, stream>>>(Sbuf, Pbuf);
    // attention head-mean -> output #2
    mean_heads_k<<<(LQ * LKV) / 256, 256, 0, stream>>>(
        Sbuf, out_mean + (size_t)b * LQ * LKV);
    // ctx: ctxb[b*LQ + q, h*HD + hd] = P_h @ V_h  (small kernel: N=64)
    gemm_wmma_small_k<<<dim3(HD / 64, LQ / 64, NH), dim3(128), 0, stream>>>(
        Pbuf, Vb + (size_t)b * LKV * DM, nullptr,
        ctxb + (size_t)b * LQ * DM,
        LKV, LKV, DM, DM,
        /*Abs*/ (long long)LQ * LKV, /*Bbs*/ HD, /*Cbs*/ HD,
        1.f, /*outF16*/ 1);
  }

  // 6) out-proj + residual: xbuf = tight + (ctxb @ Wo^T + bo) * clip(cgam)
  gemm_big(stream, ctxb, WtOut, out_b, xbuf, false, tight, cgam,
           (int)MQ, DM, DM, DM, DM, DM, 0, 0, 0, 1.f, 2, 1);

  // 7) final LN -> f16
  ln_rows_k<<<MQ, 256, 0, stream>>>(xbuf, lnf, ln_f_g, ln_f_b, LQ, LQ, 0);

  // 8) FFN1 with exact GELU (f16 out)
  gemm_big(stream, lnf, WtW1, b1, h1, true, nullptr, nullptr,
           (int)MQ, DFF, DM, DM, DFF, DFF, 0, 0, 0, 1.f, 1, 1);

  // 9) FFN2 + residual straight into d_out: x = xbuf + (h1 @ W2^T + b2) * clip(fgam)
  gemm_big(stream, h1, WtW2, b2, out_x, false, xbuf, fgam,
           (int)MQ, DM, DFF, DFF, DM, DM, 0, 0, 0, 1.f, 2, 1);
}
